// Model_18846316495445
// MI455X (gfx1250) — matrix operand order repaired and checked by reading, not compiled, not run on hardware
//
#include <hip/hip_runtime.h>
#include <hip/hip_bf16.h>
#include <math.h>

// ---------------- model constants ----------------
constexpr int NB   = 32;    // batch
constexpr int NS   = 512;   // seq len
constexpr int NC   = 38;    // channels (enc_in / c_out)
constexpr int ND   = 512;   // d_model
constexpr int NH   = 8;     // heads
constexpr int NDK  = 64;    // head dim
constexpr int NDFF = 2048;  // d_ff
constexpr int NT   = NB * NS; // tokens = 16384
constexpr int PH0  = 128, PH1 = 128;

typedef __attribute__((ext_vector_type(16))) __bf16 v16bf;
typedef __attribute__((ext_vector_type(8)))  __bf16 v8bf;
typedef __attribute__((ext_vector_type(8)))  float  v8f;

static __device__ inline __bf16 f2bf(float f) {
  union { float f; unsigned u; } v; v.f = f;
  unsigned r = v.u + 0x7FFFu + ((v.u >> 16) & 1u);   // round-to-nearest-even
  unsigned short h = (unsigned short)(r >> 16);
  __bf16 b; __builtin_memcpy(&b, &h, 2); return b;
}

static __device__ inline v16bf cat8(v8bf lo, v8bf hi) {
  return __builtin_shufflevector(lo, hi, 0, 1, 2, 3, 4, 5, 6, 7,
                                 8, 9, 10, 11, 12, 13, 14, 15);
}

static __device__ inline float gelu_exact(float x) {
  return 0.5f * x * (1.0f + erff(x * 0.70710678118654752f));
}

// ---------------- per-(b,c) stats + normalization ----------------
__global__ void k_stats(const float* __restrict__ x, float* xn, float* meanv, float* stdv) {
  int idx = blockIdx.x * blockDim.x + threadIdx.x;
  if (idx >= NB * NC) return;
  int b = idx / NC, c = idx % NC;
  float s = 0.f;
  for (int t = 0; t < NS; ++t) s += x[(b * NS + t) * NC + c];
  float m = s / NS;
  float v = 0.f;
  for (int t = 0; t < NS; ++t) { float d = x[(b * NS + t) * NC + c] - m; v += d * d; }
  v /= NS;
  float sd = sqrtf(v + 1e-5f);
  meanv[idx] = m; stdv[idx] = sd;
  float inv = 1.f / sd;
  for (int t = 0; t < NS; ++t)
    xn[(b * NS + t) * NC + c] = (x[(b * NS + t) * NC + c] - m) * inv;
}

// ---------------- projector: series conv (win as channels) + concat stats ----------------
__global__ void k_projcat(const float* __restrict__ x, const float* __restrict__ Wc,
                          const float* __restrict__ stats, float* cat) {
  int b = blockIdx.x;
  for (int c = threadIdx.x; c < NC; c += blockDim.x) {
    int c0 = (c - 1 + NC) % NC, c2 = (c + 1) % NC;
    float acc = 0.f;
    for (int s = 0; s < NS; ++s) {
      const float* w  = Wc + s * 3;
      const float* xr = x + (size_t)(b * NS + s) * NC;
      acc += w[0] * xr[c0] + w[1] * xr[c] + w[2] * xr[c2];
    }
    cat[b * (2 * NC) + c]      = acc;
    cat[b * (2 * NC) + NC + c] = stats[b * NC + c];
  }
}

// ---------------- projector MLP: 76 -> 128 -> 128 -> out_dim ----------------
__global__ void k_mlp(const float* __restrict__ cat,
                      const float* __restrict__ w0, const float* __restrict__ b0,
                      const float* __restrict__ w1, const float* __restrict__ b1,
                      const float* __restrict__ wo, float* out, int out_dim, int do_exp) {
  int b = blockIdx.x, j = threadIdx.x; // blockDim = 128
  __shared__ float vin[2 * NC];
  __shared__ float h1[PH0];
  __shared__ float h2[PH1];
  if (j < 2 * NC) vin[j] = cat[b * 2 * NC + j];
  __syncthreads();
  float a = b0[j];
  for (int i = 0; i < 2 * NC; ++i) a += vin[i] * w0[i * PH0 + j];
  h1[j] = fmaxf(a, 0.f);
  __syncthreads();
  float a2 = b1[j];
  for (int i = 0; i < PH0; ++i) a2 += h1[i] * w1[i * PH1 + j];
  h2[j] = fmaxf(a2, 0.f);
  __syncthreads();
  for (int o = j; o < out_dim; o += blockDim.x) {
    float acc = 0.f;
    for (int i = 0; i < PH1; ++i) acc += h2[i] * wo[i * out_dim + o];
    out[b * out_dim + o] = do_exp ? __expf(acc) : acc;
  }
}

// ---------------- token conv embedding (circular k=3) + sinusoidal PE ----------------
__global__ void k_embed(const float* __restrict__ x, const float* __restrict__ W, float* out) {
  int bs = blockIdx.x; int b = bs / NS; int s = bs % NS;
  __shared__ float rows[3 * NC];
  for (int i = threadIdx.x; i < 3 * NC; i += blockDim.x) {
    int k = i / NC, c = i % NC;
    int sp = (s + k - 1 + NS) % NS;
    rows[i] = x[(size_t)(b * NS + sp) * NC + c];
  }
  __syncthreads();
  for (int d = threadIdx.x; d < ND; d += blockDim.x) {
    float acc = 0.f;
    for (int c = 0; c < NC; ++c)
      for (int k = 0; k < 3; ++k)
        acc += W[(d * NC + c) * 3 + k] * rows[k * NC + c];
    int p = d >> 1;
    float freq = __expf(-(float)(2 * p) * (logf(10000.f) / (float)ND));
    float ang  = (float)s * freq;
    acc += (d & 1) ? __cosf(ang) : __sinf(ang);
    out[(size_t)bs * ND + d] = acc;
  }
}

// ---------------- WMMA GEMM: C[M,N] = act(A[M,K] @ W + bias) ----------------
// W row-major [K,N] if transW==0, else W stored [N,K] (use W^T).
// block = 128 thr (4 waves), 64x64 C tile, K step 32, bf16 in / f32 acc.
// LDS layouts chosen so each lane's WMMA fragment is contiguous 16B chunks
// (ds_load_b128), A: row-major stride 48 (96B), B: transposed n-major stride 48.
__global__ void k_gemm(const float* __restrict__ A, const float* __restrict__ W,
                       const float* __restrict__ bias, float* __restrict__ Cm,
                       int M, int N, int K, int transW, int act) {
  constexpr int ASR = 48, BSR = 48;
  __shared__ __bf16 As[64 * ASR];      // [row 64][k 32]
  __shared__ __bf16 Bt[64 * BSR];      // [n 64][k 32] (transposed)
  const int tid  = threadIdx.x;
  const int lane = tid & 31, wv = tid >> 5;
  const int m = lane & 15, half = lane >> 4;
  const int n0 = blockIdx.x * 64, m0 = blockIdx.y * 64;

  v8f acc[4];
  for (int t = 0; t < 4; ++t)
    for (int r = 0; r < 8; ++r) acc[t][r] = 0.f;

  for (int k0 = 0; k0 < K; k0 += 32) {
    // stage A (coalesced along k)
    for (int i = tid; i < 64 * 32; i += 128) {
      int r = i >> 5, c = i & 31;
      As[r * ASR + c] = f2bf(A[(size_t)(m0 + r) * K + k0 + c]);
    }
    // stage B transposed; pick coalesced mapping per layout
    if (transW) {
      for (int i = tid; i < 64 * 32; i += 128) {
        int r = i & 31, c = i >> 5;   // consecutive tid -> consecutive k
        Bt[c * BSR + r] = f2bf(W[(size_t)(n0 + c) * K + (k0 + r)]);
      }
    } else {
      for (int i = tid; i < 64 * 32; i += 128) {
        int c = i & 63, r = i >> 6;   // consecutive tid -> consecutive n
        Bt[c * BSR + r] = f2bf(W[(size_t)(k0 + r) * N + (n0 + c)]);
      }
    }
    // prefetch next K tile (global_prefetch_b8 path)
    if (k0 + 32 < K) {
      __builtin_prefetch(&A[(size_t)(m0 + (tid & 63)) * K + k0 + 32], 0, 1);
      if (transW) __builtin_prefetch(&W[(size_t)(n0 + (tid & 63)) * K + k0 + 32], 0, 1);
      else        __builtin_prefetch(&W[(size_t)(k0 + 32 + (tid & 31)) * N + n0], 0, 1);
    }
    __syncthreads();

    const int arow = (wv * 16 + m) * ASR;
    v16bf a = cat8(*(const v8bf*)&As[arow + (half << 4)],
                   *(const v8bf*)&As[arow + 8 + (half << 4)]);
#pragma unroll
    for (int t = 0; t < 4; ++t) {
      const v8bf* bp = (const v8bf*)&Bt[(t * 16 + m) * BSR + (half << 4)];
      v16bf bf = cat8(bp[0], bp[1]);
      acc[t] = __builtin_amdgcn_wmma_f32_16x16x32_bf16(
          false, a, false, bf, (short)0, acc[t], false, false);
    }
    __syncthreads();
  }

  for (int t = 0; t < 4; ++t) {
    int col = n0 + t * 16 + m;
    float bv = bias ? bias[col] : 0.f;
    for (int r = 0; r < 8; ++r) {
      float x = acc[t][r] + bv;
      if (act == 1) x = gelu_exact(x);
      Cm[(size_t)(m0 + wv * 16 + half * 8 + r) * N + col] = x;
    }
  }
}

// ---------------- fused DS-attention: one wave per (b, h, 16-row l tile) ----------------
__global__ void k_attn(const float* __restrict__ Qp, const float* __restrict__ Kp,
                       const float* __restrict__ Vp, float* __restrict__ Op,
                       const float* __restrict__ tau, const float* __restrict__ delta,
                       int causal) {
  constexpr int SP = NS + 8;
  __shared__ float sc[16][SP];
  __shared__ __bf16 Vt[64 * 40];          // V tile transposed: [col 64][row 32]
  const int blk = blockIdx.x;
  const int lt = blk & 31;                // S/16 = 32 tiles
  const int h  = (blk >> 5) & 7;
  const int b  = blk >> 8;
  const int l0 = lt * 16;
  const int lane = threadIdx.x, m = lane & 15, half = lane >> 4;
  const int ho = h * NDK;
  const float tb = tau[b];

  // Q fragments (16x64 tile; K-dim 64 -> 2 chunks of 32); contiguous float4 loads
  v16bf aq[2];
#pragma unroll
  for (int c = 0; c < 2; ++c) {
    const float* qr = Qp + (size_t)(b * NS + l0 + m) * ND + ho + c * 32;
    int e0 = half << 4;
    float4 qa = *(const float4*)(qr + e0);
    float4 qb = *(const float4*)(qr + e0 + 4);
    float4 qc = *(const float4*)(qr + e0 + 8);
    float4 qd = *(const float4*)(qr + e0 + 12);
    aq[c][0] = f2bf(qa.x); aq[c][1] = f2bf(qa.y); aq[c][2] = f2bf(qa.z); aq[c][3] = f2bf(qa.w);
    aq[c][4] = f2bf(qb.x); aq[c][5] = f2bf(qb.y); aq[c][6] = f2bf(qb.z); aq[c][7] = f2bf(qb.w);
    aq[c][8]  = f2bf(qc.x); aq[c][9]  = f2bf(qc.y); aq[c][10] = f2bf(qc.z); aq[c][11] = f2bf(qc.w);
    aq[c][12] = f2bf(qd.x); aq[c][13] = f2bf(qd.y); aq[c][14] = f2bf(qd.z); aq[c][15] = f2bf(qd.w);
  }

  // scores = (Q K^T * tau + delta) / sqrt(dk), causal mask -> LDS
  for (int s0 = 0; s0 < NS; s0 += 16) {
    v8f acc;
    for (int r = 0; r < 8; ++r) acc[r] = 0.f;
#pragma unroll
    for (int c = 0; c < 2; ++c) {
      const float* kr = Kp + (size_t)(b * NS + s0 + m) * ND + ho + c * 32;
      int e0 = half << 4;               // B frag: 16 contiguous K values
      float4 ka = *(const float4*)(kr + e0);
      float4 kb = *(const float4*)(kr + e0 + 4);
      float4 kc = *(const float4*)(kr + e0 + 8);
      float4 kd = *(const float4*)(kr + e0 + 12);
      v16bf bk;
      bk[0] = f2bf(ka.x); bk[1] = f2bf(ka.y); bk[2] = f2bf(ka.z); bk[3] = f2bf(ka.w);
      bk[4] = f2bf(kb.x); bk[5] = f2bf(kb.y); bk[6] = f2bf(kb.z); bk[7] = f2bf(kb.w);
      bk[8]  = f2bf(kc.x); bk[9]  = f2bf(kc.y); bk[10] = f2bf(kc.z); bk[11] = f2bf(kc.w);
      bk[12] = f2bf(kd.x); bk[13] = f2bf(kd.y); bk[14] = f2bf(kd.z); bk[15] = f2bf(kd.w);
      acc = __builtin_amdgcn_wmma_f32_16x16x32_bf16(
          false, aq[c], false, bk, (short)0, acc, false, false);
    }
    int col = s0 + m;
    float dv = delta ? delta[b * NS + col] : 0.f;
    for (int r = 0; r < 8; ++r) {
      int row = half * 8 + r;
      float vsc = (acc[r] * tb + dv) * 0.125f;   // 1/sqrt(64)
      if (causal && col > l0 + row) vsc = -1e30f;
      sc[row][col] = vsc;
    }
  }

  // softmax over each of 16 rows (whole wave per row)
  for (int r = 0; r < 16; ++r) {
    float mx = -1e30f;
    for (int j = lane; j < NS; j += 32) mx = fmaxf(mx, sc[r][j]);
    for (int o = 16; o; o >>= 1) mx = fmaxf(mx, __shfl_xor(mx, o, 32));
    float sum = 0.f;
    for (int j = lane; j < NS; j += 32) {
      float e = __expf(sc[r][j] - mx);
      sc[r][j] = e; sum += e;
    }
    for (int o = 16; o; o >>= 1) sum += __shfl_xor(sum, o, 32);
    float inv = 1.f / sum;
    for (int j = lane; j < NS; j += 32) sc[r][j] *= inv;
  }

  // O = P @ V  (K-dim = S, step 32; 4 N-tiles of 16 cover dk=64)
  v8f oacc[4];
  for (int t = 0; t < 4; ++t)
    for (int r = 0; r < 8; ++r) oacc[t][r] = 0.f;
  for (int s0 = 0; s0 < NS; s0 += 32) {
    // stage V tile [s0..s0+32) x [ho..ho+64) transposed to bf16 in LDS
    for (int j = lane; j < 32 * 64; j += 32) {
      int c = j & 63, r = j >> 6;      // consecutive lanes -> consecutive cols
      Vt[c * 40 + r] = f2bf(Vp[(size_t)(b * NS + s0 + r) * ND + ho + c]);
    }
    // P fragment: contiguous float4 loads from score LDS
    const float* pr = &sc[m][s0];
    int e0 = half << 4;
    float4 pa = *(const float4*)(pr + e0);
    float4 pb = *(const float4*)(pr + e0 + 4);
    float4 pc = *(const float4*)(pr + e0 + 8);
    float4 pd = *(const float4*)(pr + e0 + 12);
    v16bf ap;
    ap[0] = f2bf(pa.x); ap[1] = f2bf(pa.y); ap[2] = f2bf(pa.z); ap[3] = f2bf(pa.w);
    ap[4] = f2bf(pb.x); ap[5] = f2bf(pb.y); ap[6] = f2bf(pb.z); ap[7] = f2bf(pb.w);
    ap[8]  = f2bf(pc.x); ap[9]  = f2bf(pc.y); ap[10] = f2bf(pc.z); ap[11] = f2bf(pc.w);
    ap[12] = f2bf(pd.x); ap[13] = f2bf(pd.y); ap[14] = f2bf(pd.z); ap[15] = f2bf(pd.w);
#pragma unroll
    for (int t = 0; t < 4; ++t) {
      const v8bf* vp = (const v8bf*)&Vt[(t * 16 + m) * 40 + (half << 4)];
      v16bf bv = cat8(vp[0], vp[1]);
      oacc[t] = __builtin_amdgcn_wmma_f32_16x16x32_bf16(
          false, ap, false, bv, (short)0, oacc[t], false, false);
    }
  }
  for (int t = 0; t < 4; ++t)
    for (int r = 0; r < 8; ++r)
      Op[(size_t)(b * NS + l0 + half * 8 + r) * ND + ho + t * 16 + m] = oacc[t][r];
}

// ---------------- series decomposition (MA=25, replicate pad) ----------------
__global__ void k_decomp(const float* __restrict__ a, const float* __restrict__ bsrc,
                         float* __restrict__ seas, float* __restrict__ trend,
                         int mode, int S, int Ch, int total) {
  int idx = blockIdx.x * blockDim.x + threadIdx.x;
  if (idx >= total) return;
  int c = idx % Ch, s = (idx / Ch) % S, bb = idx / (Ch * S);
  float acc = 0.f;
  for (int j = 0; j < 25; ++j) {
    int sp = s + j - 12;
    sp = sp < 0 ? 0 : (sp >= S ? S - 1 : sp);
    size_t id2 = (size_t)(bb * S + sp) * Ch + c;
    float v = a[id2]; if (bsrc) v += bsrc[id2];
    acc += v;
  }
  float tr = acc * (1.0f / 25.0f);
  float x = a[idx]; if (bsrc) x += bsrc[idx];
  seas[idx] = x - tr;
  if (mode == 1) trend[idx] = tr;
  else if (mode == 2) trend[idx] += tr;
}

// ---------------- layernorm over D ----------------
__global__ void k_ln(const float* __restrict__ x, const float* __restrict__ g,
                     const float* __restrict__ bta, float* __restrict__ out) {
  int row = blockIdx.x, lane = threadIdx.x;
  const float* xr = x + (size_t)row * ND;
  float s = 0.f;
  for (int d = lane; d < ND; d += 32) s += xr[d];
  for (int o = 16; o; o >>= 1) s += __shfl_xor(s, o, 32);
  float m = s / ND;
  float v = 0.f;
  for (int d = lane; d < ND; d += 32) { float dd = xr[d] - m; v += dd * dd; }
  for (int o = 16; o; o >>= 1) v += __shfl_xor(v, o, 32);
  float inv = rsqrtf(v / ND + 1e-5f);
  for (int d = lane; d < ND; d += 32)
    out[(size_t)row * ND + d] = (xr[d] - m) * inv * g[d] + bta[d];
}

// ---------------- decoder trend conv (circular k=3, D->C), trend_c += ----------------
__global__ void k_trendconv(const float* __restrict__ t, const float* __restrict__ tp,
                            float* __restrict__ trend_c) {
  int bs = blockIdx.x; int b = bs / NS; int s = bs % NS;
  __shared__ float rows[3 * ND];
  for (int i = threadIdx.x; i < 3 * ND; i += blockDim.x) {
    int k = i / ND, d = i % ND;
    int sp = (s + k - 1 + NS) % NS;
    rows[i] = t[(size_t)(b * NS + sp) * ND + d];
  }
  __syncthreads();
  for (int c = threadIdx.x; c < NC; c += blockDim.x) {
    float acc = 0.f;
    for (int k = 0; k < 3; ++k)
      for (int d = 0; d < ND; ++d)
        acc += tp[(size_t)(c * ND + d) * 3 + k] * rows[k * ND + d];
    trend_c[(size_t)bs * NC + c] += acc;
  }
}

// ---------------- final: proj + trend, de-normalize ----------------
__global__ void k_final(const float* __restrict__ x, const float* __restrict__ Wp,
                        const float* __restrict__ bp, const float* __restrict__ trend,
                        const float* __restrict__ stdv, const float* __restrict__ meanv,
                        float* __restrict__ out) {
  int row = blockIdx.x; int b = row / NS;
  __shared__ float xr[ND];
  for (int d = threadIdx.x; d < ND; d += blockDim.x) xr[d] = x[(size_t)row * ND + d];
  __syncthreads();
  for (int c = threadIdx.x; c < NC; c += blockDim.x) {
    float acc = bp[c];
    for (int d = 0; d < ND; ++d) acc += xr[d] * Wp[d * NC + c];
    out[(size_t)row * NC + c] =
        (acc + trend[(size_t)row * NC + c]) * stdv[b * NC + c] + meanv[b * NC + c];
  }
}

// ================= host orchestration =================
extern "C" void kernel_launch(void* const* d_in, const int* in_sizes, int n_in,
                              void* d_out, int out_size, void* d_ws, size_t ws_size,
                              hipStream_t stream) {
  (void)in_sizes; (void)n_in; (void)out_size; (void)ws_size;
  auto P = [&](int i) { return (const float*)d_in[i]; };

  // input leaves in setup_inputs() insertion order
  const float* x_enc = P(0);
  const float *tconv = P(1), *tw0 = P(2), *tw1 = P(3), *tb0 = P(4), *tb1 = P(5), *tout = P(6);
  const float *dconv = P(7), *dw0 = P(8), *dw1 = P(9), *db0 = P(10), *db1 = P(11), *dout_w = P(12);
  const float *enc_emb = P(13), *dec_emb = P(14);
  const int EB[2] = {15, 25};  // enc layers: Wq,Wk,Wv,Wo,bq,bk,bv,bo,c1,c2
  const float *enc_g = P(35), *enc_b = P(36);
  const int SB = 37, CB = 45;  // decoder self / cross attn bases
  const float *dc1 = P(53), *dc2 = P(54), *dtp = P(55);
  const float *dec_g = P(56), *dec_b = P(57);
  const float *projW = P(58), *projB = P(59);

  // workspace layout (floats)
  float* ws = (float*)d_ws;
  size_t off = 0;
  auto alloc = [&](size_t n) { float* p = ws + off; off += n; return p; };
  float* XN    = alloc((size_t)NT * NC);
  float* SEASC = alloc((size_t)NT * NC);
  float* TRC   = alloc((size_t)NT * NC);
  float* MEANV = alloc(NB * NC);
  float* STDV  = alloc(NB * NC);
  float* TAU   = alloc(NB);
  float* DELTA = alloc(NB * NS);
  float* CATT  = alloc(NB * 2 * NC);
  float* CATD  = alloc(NB * 2 * NC);
  float* X0    = alloc((size_t)NT * ND);
  float* X1    = alloc((size_t)NT * ND);
  float* Yb    = alloc((size_t)NT * ND);
  float* Qb    = alloc((size_t)NT * ND);
  float* Kb    = alloc((size_t)NT * ND);
  float* Vb    = alloc((size_t)NT * ND);
  float* CTX   = alloc((size_t)NT * ND);
  float* ENC   = alloc((size_t)NT * ND);
  float* TSUM  = alloc((size_t)NT * ND);
  float* HID   = alloc((size_t)NT * NDFF);

  auto gemm = [&](const float* A, const float* W, const float* bias, float* Cm,
                  int M, int N, int K, int tw, int act) {
    dim3 g(N / 64, M / 64, 1);
    k_gemm<<<g, dim3(128), 0, stream>>>(A, W, bias, Cm, M, N, K, tw, act);
  };
  auto decomp = [&](const float* a, const float* b, float* seas, float* tr,
                    int mode, int Ch) {
    int total = NB * NS * Ch;
    k_decomp<<<(total + 255) / 256, 256, 0, stream>>>(a, b, seas, tr, mode, NS, Ch, total);
  };
  auto attn = [&](const float* Q, const float* K, const float* V, float* O,
                  const float* delta, int causal) {
    k_attn<<<NB * NH * (NS / 16), 32, 0, stream>>>(Q, K, V, O, TAU, delta, causal);
  };

  // 1) stats + normalization
  k_stats<<<(NB * NC + 63) / 64, 64, 0, stream>>>(x_enc, XN, MEANV, STDV);

  // 2) tau / delta projectors (operate on raw x + stats)
  k_projcat<<<NB, 64, 0, stream>>>(x_enc, tconv, STDV, CATT);
  k_mlp<<<NB, 128, 0, stream>>>(CATT, tw0, tb0, tw1, tb1, tout, TAU, 1, 1);
  k_projcat<<<NB, 64, 0, stream>>>(x_enc, dconv, MEANV, CATD);
  k_mlp<<<NB, 128, 0, stream>>>(CATD, dw0, db0, dw1, db1, dout_w, DELTA, NS, 0);

  // 3) encoder embedding
  k_embed<<<NT, 128, 0, stream>>>(XN, enc_emb, X0);

  // 4) encoder layers
  for (int l = 0; l < 2; ++l) {
    int b0i = EB[l];
    gemm(X0, P(b0i + 0), P(b0i + 4), Qb, NT, ND, ND, 0, 0);   // Q
    gemm(X0, P(b0i + 1), P(b0i + 5), Kb, NT, ND, ND, 0, 0);   // K
    gemm(X0, P(b0i + 2), P(b0i + 6), Vb, NT, ND, ND, 0, 0);   // V
    attn(Qb, Kb, Vb, CTX, DELTA, 0);
    gemm(CTX, P(b0i + 3), P(b0i + 7), Yb, NT, ND, ND, 0, 0);  // O proj
    decomp(X0, Yb, X1, nullptr, 0, ND);
    gemm(X1, P(b0i + 8), nullptr, HID, NT, NDFF, ND, 1, 1);   // x @ c1.T, gelu
    gemm(HID, P(b0i + 9), nullptr, Yb, NT, ND, NDFF, 1, 0);   // h @ c2.T
    decomp(X1, Yb, X0, nullptr, 0, ND);
  }
  k_ln<<<NT, 32, 0, stream>>>(X0, enc_g, enc_b, ENC);

  // 5) decoder input: decomp(x_n) -> seasonal/trend, embed seasonal
  decomp(XN, nullptr, SEASC, TRC, 1, NC);
  k_embed<<<NT, 128, 0, stream>>>(SEASC, dec_emb, X0);

  // 6) decoder layer
  gemm(X0, P(SB + 0), P(SB + 4), Qb, NT, ND, ND, 0, 0);
  gemm(X0, P(SB + 1), P(SB + 5), Kb, NT, ND, ND, 0, 0);
  gemm(X0, P(SB + 2), P(SB + 6), Vb, NT, ND, ND, 0, 0);
  attn(Qb, Kb, Vb, CTX, nullptr, 1);                          // causal self-attn
  gemm(CTX, P(SB + 3), P(SB + 7), Yb, NT, ND, ND, 0, 0);
  decomp(X0, Yb, X1, TSUM, 1, ND);                            // t1

  gemm(X1, P(CB + 0), P(CB + 4), Qb, NT, ND, ND, 0, 0);
  gemm(ENC, P(CB + 1), P(CB + 5), Kb, NT, ND, ND, 0, 0);
  gemm(ENC, P(CB + 2), P(CB + 6), Vb, NT, ND, ND, 0, 0);
  attn(Qb, Kb, Vb, CTX, nullptr, 0);                          // cross-attn
  gemm(CTX, P(CB + 3), P(CB + 7), Yb, NT, ND, ND, 0, 0);
  decomp(X1, Yb, X0, TSUM, 2, ND);                            // += t2

  gemm(X0, dc1, nullptr, HID, NT, NDFF, ND, 1, 1);
  gemm(HID, dc2, nullptr, Yb, NT, ND, NDFF, 1, 0);
  decomp(X0, Yb, X1, TSUM, 2, ND);                            // += t3

  k_trendconv<<<NT, 128, 0, stream>>>(TSUM, dtp, TRC);        // trend += conv(t1+t2+t3)

  // 7) final norm, projection, de-normalize
  k_ln<<<NT, 32, 0, stream>>>(X1, dec_g, dec_b, X0);
  k_final<<<NT, 64, 0, stream>>>(X0, projW, projB, TRC, STDV, MEANV, (float*)d_out);
}
